// ToSTBlock_15401752723501
// MI455X (gfx1250) — matrix operand order repaired and checked by reading, not compiled, not run on hardware
//
#include <hip/hip_runtime.h>
#include <hip/hip_bf16.h>
#include <cstdint>
#include <cstddef>

// ---------------------------------------------------------------------------
// Types for CDNA5 WMMA (wave32): v16bf operands, v8f accumulator.
// ---------------------------------------------------------------------------
typedef __bf16 bf16_t;
typedef __attribute__((ext_vector_type(16))) __bf16 v16bf;
typedef __attribute__((ext_vector_type(8)))  __bf16 v8bf;
typedef __attribute__((ext_vector_type(8)))  float  v8f;
typedef int v4i32 __attribute__((vector_size(16)));   // matches async-LDS builtin param

// Problem constants (from reference setup_inputs).
constexpr int BB  = 16;    // batch
constexpr int NT  = 1024;  // tokens = H*W = 32*32
constexpr int CC  = 1024;  // channels
constexpr int DFF = 4096;  // MLP hidden
constexpr int NH  = 8;     // heads
constexpr int DH  = 128;   // head dim
constexpr float EPS16 = 0.0009765625f;
constexpr float LNEPS = 1e-5f;

// CDNA5 async global->LDS copy (ASYNCcnt-tracked, bypasses VGPRs).
#if defined(__AMDGCN__) && __has_builtin(__builtin_amdgcn_global_load_async_to_lds_b128)
#define HAS_ASYNC_LDS 1
#else
#define HAS_ASYNC_LDS 0
#endif

__device__ __forceinline__ void wait_async_zero()
{
#if defined(__AMDGCN__)
#if __has_builtin(__builtin_amdgcn_s_wait_asynccnt)
    __builtin_amdgcn_s_wait_asynccnt(0);
#else
    asm volatile("s_wait_asynccnt 0" ::: "memory");
#endif
#endif
}

// ---------------------------------------------------------------------------
// fp32 -> bf16 weight conversion
// ---------------------------------------------------------------------------
__global__ void __launch_bounds__(256)
f32_to_bf16_kernel(const float* __restrict__ in, bf16_t* __restrict__ out, int n)
{
    int i = blockIdx.x * 256 + threadIdx.x;
    if (i < n) out[i] = (bf16_t)in[i];
}

// ---------------------------------------------------------------------------
// LayerNorm over C; input in (B, C, N) layout (the 4d image layout), output
// bf16 in (B, N, C) layout ready to be the WMMA A operand.
// One block per (b, n); 256 threads, 4 channels each.
// ---------------------------------------------------------------------------
__global__ void __launch_bounds__(256)
layernorm_bf16_kernel(const float* __restrict__ x4, const float* __restrict__ w,
                      const float* __restrict__ bta, bf16_t* __restrict__ out3)
{
    const int bn = blockIdx.x;              // b*NT + n
    const int b  = bn / NT;
    const int n  = bn % NT;
    const float* xb = x4 + (size_t)b * CC * NT + n;

    float vals[4];
    float s = 0.f, s2 = 0.f;
#pragma unroll
    for (int i = 0; i < 4; ++i) {
        int c = threadIdx.x + i * 256;
        float v = xb[(size_t)c * NT];
        vals[i] = v;
        s += v;
        s2 += v * v;
    }
    // wave reduce (wave32) then cross-wave via LDS
    for (int o = 16; o > 0; o >>= 1) {
        s  += __shfl_xor(s,  o, 32);
        s2 += __shfl_xor(s2, o, 32);
    }
    __shared__ float red[2][8];
    const int lane = threadIdx.x & 31;
    const int wid  = threadIdx.x >> 5;
    if (lane == 0) { red[0][wid] = s; red[1][wid] = s2; }
    __syncthreads();
    float ts = 0.f, ts2 = 0.f;
#pragma unroll
    for (int j = 0; j < 8; ++j) { ts += red[0][j]; ts2 += red[1][j]; }
    const float mu   = ts / (float)CC;
    const float var  = ts2 / (float)CC - mu * mu;   // biased variance
    const float rstd = rsqrtf(var + LNEPS);

    bf16_t* ob = out3 + (size_t)bn * CC;
#pragma unroll
    for (int i = 0; i < 4; ++i) {
        int c = threadIdx.x + i * 256;
        ob[c] = (bf16_t)((vals[i] - mu) * rstd * w[c] + bta[c]);
    }
}

// ---------------------------------------------------------------------------
// Tiled bf16 WMMA GEMM:  D[b, m, o] = sum_k A[b, m, k] * W[o, k]  (+ epilogue)
//
// Block: 256 threads = 8 waves; tile 128(M) x 128(O); K step 32.
// Wave grid 4x2: wave owns 32 rows x 64 cols = 2x4 accumulators of 16x16.
// LDS double-buffered, padded row stride (40 bf16 = 80 B -> 16B-aligned
// ds_load_b128, conflict-free across the 16 rows a half-wave touches).
// Tiles are filled with GLOBAL_LOAD_ASYNC_TO_LDS_B128 when available, so the
// HBM->LDS transfer of tile k+1 overlaps the WMMA work on tile k; each wave
// drains its ASYNCcnt before the workgroup barrier.
//
// MODE 0: store fp32 (B, M, O)
// MODE 2: + bias, exact GELU, store bf16 (B, M, O)
// MODE 3: + optional bias, out4[b,o,m] = res4[b,o,m] + gamma[o]*val  (fp32)
// ---------------------------------------------------------------------------
template<int MODE>
__global__ void __launch_bounds__(256)
gemm_bf16_wmma(const bf16_t* __restrict__ A, const bf16_t* __restrict__ W,
               float* __restrict__ outF, bf16_t* __restrict__ outH,
               const float* __restrict__ bias, const float* __restrict__ res4,
               const float* __restrict__ gamma, int M, int K, int Nout)
{
    constexpr int TM = 128, TN = 128, TK = 32, LDT = 40;
    __shared__ bf16_t As[2][TM * LDT];
    __shared__ bf16_t Bs[2][TN * LDT];

    const int tid = threadIdx.x;
    const int bz  = blockIdx.z;
    const int m0  = blockIdx.x * TM;
    const int n0  = blockIdx.y * TN;
    const bf16_t* Ab = A + (size_t)bz * M * K;

    const int lane  = tid & 31;
    const int wid   = tid >> 5;
    const int wm    = wid & 3;   // 0..3 -> row group of 32
    const int wn    = wid >> 2;  // 0..1 -> col group of 64
    const int lhalf = lane >> 4; // 0/1
    const int lrow  = lane & 15;

    v8f acc[2][4];
#pragma unroll
    for (int i = 0; i < 2; ++i)
#pragma unroll
        for (int j = 0; j < 4; ++j)
            acc[i][j] = v8f{0.f, 0.f, 0.f, 0.f, 0.f, 0.f, 0.f, 0.f};

    const int KT = K / TK;

    // Per-thread chunk coordinates: 256 threads x 2 iterations cover
    // 128 rows x 4 chunks of 16B for each of As / Bs.
    auto load_tile = [&](int kt, int buf) {
        const int k0 = kt * TK;
#if HAS_ASYNC_LDS
#pragma unroll
        for (int it = 0; it < 2; ++it) {
            int idx = it * 256 + tid;
            int row = idx >> 2;
            int ch  = idx & 3;
            const bf16_t* ga = Ab + (size_t)(m0 + row) * K + k0 + ch * 8;
            const bf16_t* gb = W  + (size_t)(n0 + row) * K + k0 + ch * 8;
            bf16_t* la = &As[buf][row * LDT + ch * 8];
            bf16_t* lb = &Bs[buf][row * LDT + ch * 8];
            __builtin_amdgcn_global_load_async_to_lds_b128(
                (v4i32*)ga, (v4i32*)la, 0, 0);
            __builtin_amdgcn_global_load_async_to_lds_b128(
                (v4i32*)gb, (v4i32*)lb, 0, 0);
        }
#else
        // Fallback: stage all chunks in distinct registers first so the
        // global loads can be issued back-to-back, then store to LDS.
        uint4 ra[2], rb[2];
#pragma unroll
        for (int it = 0; it < 2; ++it) {
            int idx = it * 256 + tid;
            int row = idx >> 2;
            int ch  = idx & 3;
            ra[it] = *(const uint4*)(Ab + (size_t)(m0 + row) * K + k0 + ch * 8);
            rb[it] = *(const uint4*)(W  + (size_t)(n0 + row) * K + k0 + ch * 8);
        }
#pragma unroll
        for (int it = 0; it < 2; ++it) {
            int idx = it * 256 + tid;
            int row = idx >> 2;
            int ch  = idx & 3;
            *(uint4*)(&As[buf][row * LDT + ch * 8]) = ra[it];
            *(uint4*)(&Bs[buf][row * LDT + ch * 8]) = rb[it];
        }
#endif
    };

    load_tile(0, 0);
#if HAS_ASYNC_LDS
    wait_async_zero();
#endif
    __syncthreads();

    for (int kt = 0; kt < KT; ++kt) {
        const int buf = kt & 1;
        if (kt + 1 < KT) load_tile(kt + 1, buf ^ 1);
        if (kt + 2 < KT) {                  // global_prefetch_b8 two tiles ahead
            const int pk = (kt + 2) * TK;
            __builtin_prefetch(Ab + (size_t)(m0 + (tid >> 1)) * K + pk, 0, 1);
            __builtin_prefetch(W  + (size_t)(n0 + (tid >> 1)) * K + pk, 0, 1);
        }

        // A fragments: 16-bit A 16x32 layout — lane<16 holds K 0..7 & 16..23,
        // lane>=16 holds K 8..15 & 24..31 for row M=lane&15.
        v16bf af[2];
#pragma unroll
        for (int tm = 0; tm < 2; ++tm) {
            const bf16_t* pa = &As[buf][(wm * 32 + tm * 16 + lrow) * LDT + lhalf * 8];
            v8bf lo = *(const v8bf*)(pa);        // K = lhalf*8 .. +7
            v8bf hi = *(const v8bf*)(pa + 16);   // K = 16 + lhalf*8 .. +7
            af[tm] = __builtin_shufflevector(lo, hi, 0, 1, 2, 3, 4, 5, 6, 7,
                                             8, 9, 10, 11, 12, 13, 14, 15);
        }
        // B fragments: lanes 0-15 hold K 0..7 & 16..23, lanes 16-31 hold K 8..15 & 24..31 for
        // column N=lane&15 (same K order as A).
        v16bf bfr[4];
#pragma unroll
        for (int tn = 0; tn < 4; ++tn) {
            const bf16_t* pb = &Bs[buf][(wn * 64 + tn * 16 + lrow) * LDT + lhalf * 8];
            v8bf lo = *(const v8bf*)(pb);
            v8bf hi = *(const v8bf*)(pb + 16);
            bfr[tn] = __builtin_shufflevector(lo, hi, 0, 1, 2, 3, 4, 5, 6, 7,
                                              8, 9, 10, 11, 12, 13, 14, 15);
        }
#pragma unroll
        for (int tm = 0; tm < 2; ++tm)
#pragma unroll
            for (int tn = 0; tn < 4; ++tn)
                acc[tm][tn] = __builtin_amdgcn_wmma_f32_16x16x32_bf16(
                    false, af[tm], false, bfr[tn], (short)0, acc[tm][tn],
                    false, false);

#if HAS_ASYNC_LDS
        wait_async_zero();      // tile kt+1 landed in LDS (overlapped WMMAs)
#endif
        __syncthreads();
    }

    // Epilogue. C/D layout: VGPR r -> M = r (+8 for lanes>=16), N = lane&15.
#pragma unroll
    for (int tm = 0; tm < 2; ++tm) {
#pragma unroll
        for (int tn = 0; tn < 4; ++tn) {
            const int row0 = m0 + wm * 32 + tm * 16 + lhalf * 8;
            const int col  = n0 + wn * 64 + tn * 16 + lrow;
            v8f a = acc[tm][tn];
#pragma unroll
            for (int r = 0; r < 8; ++r) {
                const int row = row0 + r;
                float v = a[r];
                if (MODE == 0) {
                    outF[((size_t)bz * M + row) * Nout + col] = v;
                } else if (MODE == 2) {
                    v += bias[col];
                    float g = 0.5f * v * (1.f + erff(v * 0.70710678118654752f));
                    outH[((size_t)bz * M + row) * Nout + col] = (bf16_t)g;
                } else { // MODE == 3
                    if (bias) v += bias[col];
                    size_t o = ((size_t)bz * Nout + col) * M + row;
                    outF[o] = res4[o] + gamma[col] * v;
                }
            }
        }
    }
}

// ---------------------------------------------------------------------------
// TSSA attention statistics.
// w3: (B, N, C) fp32 with C = NH*DH; head h uses slice [h*DH, (h+1)*DH).
// ---------------------------------------------------------------------------
// Phase 1: cumulative w^2 over tokens, normalized row-sum -> tmp (B,H,N).
__global__ void __launch_bounds__(128)
tssa_phase1_kernel(const float* __restrict__ w3, const float* __restrict__ db,
                   const float* __restrict__ temp, float* __restrict__ tmp)
{
    const int b = blockIdx.x, h = blockIdx.y, t = threadIdx.x; // t in [0,128)
    const float th = temp[h];
    __shared__ float red[4];
    float csum = 0.f;
    for (int n = 0; n < NT; ++n) {
        float wv = w3[((size_t)b * NT + n) * CC + h * DH + t];
        float sq = wv * wv;
        csum += sq;
        float wn = sq / fmaxf(csum, EPS16) + db[h * 1024 + n];
        float r = wn;
        for (int o = 16; o > 0; o >>= 1) r += __shfl_xor(r, o, 32);
        const int lane = t & 31, wv4 = t >> 5;
        if (lane == 0) red[wv4] = r;
        __syncthreads();
        if (t == 0)
            tmp[((size_t)b * NH + h) * NT + n] =
                (red[0] + red[1] + red[2] + red[3]) * th;
        __syncthreads();
    }
}

// Phase 2: softmax over the 8 heads for each (b, n).
__global__ void __launch_bounds__(256)
tssa_softmax_heads_kernel(const float* __restrict__ tmp, float* __restrict__ Pi)
{
    int i = blockIdx.x * 256 + threadIdx.x;   // over B*NT
    if (i >= BB * NT) return;
    const int b = i / NT, n = i % NT;
    float v[NH];
    float mx = -3.402823466e38f;
#pragma unroll
    for (int h = 0; h < NH; ++h) {
        v[h] = tmp[((size_t)b * NH + h) * NT + n];
        mx = fmaxf(mx, v[h]);
    }
    float se = 0.f;
#pragma unroll
    for (int h = 0; h < NH; ++h) { v[h] = expf(v[h] - mx); se += v[h]; }
    const float inv = 1.f / se;
#pragma unroll
    for (int h = 0; h < NH; ++h)
        Pi[((size_t)b * NH + h) * NT + n] = v[h] * inv;
}

// Phase 3: running sums of Pi and w^2*Pi, produce y = -(w*Pi)/(1+dots), bf16.
__global__ void __launch_bounds__(128)
tssa_phase3_kernel(const float* __restrict__ w3, const float* __restrict__ Pi,
                   bf16_t* __restrict__ y3)
{
    const int b = blockIdx.x, h = blockIdx.y, t = threadIdx.x;
    float cwp = 0.f, cpi = 0.f;
    for (int n = 0; n < NT; ++n) {
        float pi = Pi[((size_t)b * NH + h) * NT + n];
        float wv = w3[((size_t)b * NT + n) * CC + h * DH + t];
        float sq = wv * wv;
        cwp += sq * pi;
        cpi += pi;
        float dots = cwp / (cpi + EPS16);
        float at = 1.f / (1.f + dots);
        y3[((size_t)b * NT + n) * CC + h * DH + t] = (bf16_t)(-(wv * pi) * at);
    }
}

// ---------------------------------------------------------------------------
// Host-side orchestration.
// ---------------------------------------------------------------------------
extern "C" void kernel_launch(void* const* d_in, const int* in_sizes, int n_in,
                              void* d_out, int out_size, void* d_ws, size_t ws_size,
                              hipStream_t stream)
{
    (void)in_sizes; (void)n_in; (void)out_size; (void)ws_size;

    const float* x      = (const float*)d_in[0];   // (B, C, H, W)
    const float* ln1_w  = (const float*)d_in[1];
    const float* ln1_b  = (const float*)d_in[2];
    const float* Wattn  = (const float*)d_in[3];   // (C, C)
    const float* Wproj  = (const float*)d_in[4];   // (C, C)
    const float* temp   = (const float*)d_in[5];   // (H, 1)
    const float* dbias  = (const float*)d_in[6];   // (H, 1024, 1)
    const float* ln2_w  = (const float*)d_in[7];
    const float* ln2_b  = (const float*)d_in[8];
    const float* W1     = (const float*)d_in[9];   // (DFF, C)
    const float* b1     = (const float*)d_in[10];
    const float* W2     = (const float*)d_in[11];  // (C, DFF)
    const float* b2     = (const float*)d_in[12];
    const float* gamma1 = (const float*)d_in[13];
    const float* gamma2 = (const float*)d_in[14];
    float* out = (float*)d_out;

    // Workspace carve-up.
    char* ws = (char*)d_ws;
    size_t off = 0;
    auto take = [&](size_t bytes) -> void* {
        void* p = ws + off;
        off += (bytes + 255) & ~(size_t)255;
        return p;
    };
    bf16_t* xb3  = (bf16_t*)take((size_t)BB * NT * CC * 2);
    bf16_t* Wab  = (bf16_t*)take((size_t)CC * CC * 2);
    bf16_t* Wpb  = (bf16_t*)take((size_t)CC * CC * 2);
    bf16_t* W1b  = (bf16_t*)take((size_t)DFF * CC * 2);
    bf16_t* W2b  = (bf16_t*)take((size_t)CC * DFF * 2);
    float*  wbuf = (float*)take((size_t)BB * NT * CC * 4);
    float*  tmpb = (float*)take((size_t)BB * NH * NT * 4);
    float*  Pib  = (float*)take((size_t)BB * NH * NT * 4);
    bf16_t* ybuf = (bf16_t*)take((size_t)BB * NT * CC * 2);
    bf16_t* hbuf = (bf16_t*)take((size_t)BB * NT * DFF * 2);
    float*  xmid = (float*)take((size_t)BB * NT * CC * 4);

    // 0) weights -> bf16
    {
        int n1 = CC * CC, n2 = DFF * CC;
        f32_to_bf16_kernel<<<(n1 + 255) / 256, 256, 0, stream>>>(Wattn, Wab, n1);
        f32_to_bf16_kernel<<<(n1 + 255) / 256, 256, 0, stream>>>(Wproj, Wpb, n1);
        f32_to_bf16_kernel<<<(n2 + 255) / 256, 256, 0, stream>>>(W1, W1b, n2);
        f32_to_bf16_kernel<<<(n2 + 255) / 256, 256, 0, stream>>>(W2, W2b, n2);
    }

    // 1) LN1: x (B,C,N) -> xb3 (B,N,C) bf16
    layernorm_bf16_kernel<<<BB * NT, 256, 0, stream>>>(x, ln1_w, ln1_b, xb3);

    // 2) w = LN(x) @ Wattn^T   -> wbuf fp32 (B,N,C)
    gemm_bf16_wmma<0><<<dim3(NT / 128, CC / 128, BB), 256, 0, stream>>>(
        xb3, Wab, wbuf, nullptr, nullptr, nullptr, nullptr, NT, CC, CC);

    // 3) attention statistics
    tssa_phase1_kernel<<<dim3(BB, NH), 128, 0, stream>>>(wbuf, dbias, temp, tmpb);
    tssa_softmax_heads_kernel<<<(BB * NT + 255) / 256, 256, 0, stream>>>(tmpb, Pib);
    tssa_phase3_kernel<<<dim3(BB, NH), 128, 0, stream>>>(wbuf, Pib, ybuf);

    // 4) xmid = x + gamma1 * (y @ Wproj^T)   (written in (B,C,N) layout)
    gemm_bf16_wmma<3><<<dim3(NT / 128, CC / 128, BB), 256, 0, stream>>>(
        ybuf, Wpb, xmid, nullptr, nullptr, x, gamma1, NT, CC, CC);

    // 5) LN2: xmid -> xb3 bf16
    layernorm_bf16_kernel<<<BB * NT, 256, 0, stream>>>(xmid, ln2_w, ln2_b, xb3);

    // 6) h = gelu(LN(xmid) @ W1^T + b1)  -> bf16
    gemm_bf16_wmma<2><<<dim3(NT / 128, DFF / 128, BB), 256, 0, stream>>>(
        xb3, W1b, nullptr, hbuf, b1, nullptr, nullptr, NT, CC, DFF);

    // 7) out = xmid + gamma2 * (h @ W2^T + b2)
    gemm_bf16_wmma<3><<<dim3(NT / 128, CC / 128, BB), 256, 0, stream>>>(
        hbuf, W2b, out, nullptr, b2, xmid, gamma2, NT, DFF, CC);
}
